// CESLayer_6270652252794
// MI455X (gfx1250) — matrix operand order repaired and checked by reading, not compiled, not run on hardware
//
#include <hip/hip_runtime.h>
#include <hip/hip_bf16.h>

// ---------------------------------------------------------------------------
// out[b,o] = cos( ((x[b,:]+1)*30) . theta[o,:] + phi[o] )
// B=131072, IN=512, OUT=512, fp32 in/out.
// Memory-bound (512MB @ 23.3TB/s ~ 22us); GEMM via f16 WMMA, f32 accumulate,
// fused fast-cos epilogue (v_cos_f32 TRANS op, co-issues with VALU).
// Tiles: 128x128 per block, K-step 32, 8 waves/block, each wave = 32x64 via
// 2x4 v_wmma_f32_16x16x32_f16 accumulators. No launch-bounds occupancy cap:
// ~170 VGPRs/wave must not spill (spills cost more than occupancy here).
// ---------------------------------------------------------------------------

typedef __attribute__((ext_vector_type(16))) _Float16 v16h;
typedef __attribute__((ext_vector_type(8)))  _Float16 v8h;
typedef __attribute__((ext_vector_type(8)))  float    v8f;

#define OMEGA0   30.0f
#define IN_DIM   512
#define OUT_DIM  512
#define BATCH    131072
#define BM       128
#define BN       128
#define BK       32
#define BKP      40   // padded LDS stride (halves): 80B, 16B-aligned, bank-spread

__global__ __launch_bounds__(256)
void siren_ces_wmma(const float* __restrict__ x,
                    const float* __restrict__ theta,
                    const float* __restrict__ phi,
                    float* __restrict__ out)
{
    __shared__ _Float16 As[BM * BKP];   // x' tile, [row][k], f16
    __shared__ _Float16 Bs[BN * BKP];   // theta tile, [col][k], f16

    const int t    = threadIdx.x;
    const int lane = t & 31;
    const int wave = t >> 5;

    const int row0 = blockIdx.y * BM;   // batch rows
    const int col0 = blockIdx.x * BN;   // output cols

    // wave sub-tile: 4 waves along M (32 rows each), 2 along N (64 cols each)
    const int waveM = (wave & 3) * 32;
    const int waveN = (wave >> 2) * 64;

    v8f acc[2][4];
    #pragma unroll
    for (int a = 0; a < 2; ++a)
        #pragma unroll
        for (int b = 0; b < 4; ++b)
            acc[a][b] = (v8f){0.f,0.f,0.f,0.f,0.f,0.f,0.f,0.f};

    // tile loading assignment: 2 threads per row, 16 consecutive K each
    const int lrow  = t >> 1;          // 0..127
    const int lhalf = (t & 1) * 16;    // 0 or 16

    const int g  = lane >> 4;          // K-group (lanes 16..31 take high K)
    const int mr = lane & 15;          // M (for A) / N (for B) within 16-tile

    for (int k0 = 0; k0 < IN_DIM; k0 += BK) {
        // ---- stage x' (f32 -> (x+1)*omega0 -> f16) and theta (f32 -> f16) ----
        const float4* xp = reinterpret_cast<const float4*>(
            x + (size_t)(row0 + lrow) * IN_DIM + k0 + lhalf);
        const float4* tp = reinterpret_cast<const float4*>(
            theta + (size_t)(col0 + lrow) * IN_DIM + k0 + lhalf);
        _Float16* da = &As[lrow * BKP + lhalf];
        _Float16* db = &Bs[lrow * BKP + lhalf];
        #pragma unroll
        for (int j = 0; j < 4; ++j) {
            const float4 v = xp[j];
            da[4*j + 0] = (_Float16)fmaf(v.x, OMEGA0, OMEGA0);
            da[4*j + 1] = (_Float16)fmaf(v.y, OMEGA0, OMEGA0);
            da[4*j + 2] = (_Float16)fmaf(v.z, OMEGA0, OMEGA0);
            da[4*j + 3] = (_Float16)fmaf(v.w, OMEGA0, OMEGA0);
            const float4 w = tp[j];
            db[4*j + 0] = (_Float16)w.x;
            db[4*j + 1] = (_Float16)w.y;
            db[4*j + 2] = (_Float16)w.z;
            db[4*j + 3] = (_Float16)w.w;
        }
        // hide latency of next K-slice of x (global_prefetch_b8)
        if (k0 + BK < IN_DIM)
            __builtin_prefetch(x + (size_t)(row0 + lrow) * IN_DIM + k0 + BK + lhalf, 0, 1);

        __syncthreads();

        // ---- gather WMMA fragments (ISA 7.12.2 16-bit layouts) ----
        // A 16x32: halves 0..7 = K g*8+0..7 ; halves 8..15 = K 16+g*8+0..7
        v16h af[2];
        #pragma unroll
        for (int a = 0; a < 2; ++a) {
            const _Float16* base = &As[(waveM + a * 16 + mr) * BKP];
            const v8h lo = *reinterpret_cast<const v8h*>(base + g * 8);
            const v8h hi = *reinterpret_cast<const v8h*>(base + 16 + g * 8);
            #pragma unroll
            for (int j = 0; j < 8; ++j) { af[a][j] = lo[j]; af[a][8 + j] = hi[j]; }
        }
        // B 32x16: lane = column N, halves 0..7 = K g*8+0..7 ; halves 8..15 = K 16+g*8+0..7
        v16h bf[4];
        #pragma unroll
        for (int b = 0; b < 4; ++b) {
            const _Float16* base = &Bs[(waveN + b * 16 + mr) * BKP];
            const v8h lo = *reinterpret_cast<const v8h*>(base + g * 8);
            const v8h hi = *reinterpret_cast<const v8h*>(base + 16 + g * 8);
            #pragma unroll
            for (int j = 0; j < 8; ++j) { bf[b][j] = lo[j]; bf[b][8 + j] = hi[j]; }
        }

        // ---- 8 WMMAs: D = A x B + C ----
        #pragma unroll
        for (int a = 0; a < 2; ++a)
            #pragma unroll
            for (int b = 0; b < 4; ++b)
                acc[a][b] = __builtin_amdgcn_wmma_f32_16x16x32_f16(
                    false, af[a], false, bf[b],
                    (short)0, acc[a][b], false, false);

        __syncthreads();
    }

    // ---- fused epilogue: cos(phase + phi) via v_cos_f32, fp32 stores ----
    // C layout: VGPR r -> M = r + (lane>=16 ? 8 : 0), N = lane&15
    #pragma unroll
    for (int b = 0; b < 4; ++b) {
        const int n  = col0 + waveN + b * 16 + mr;
        const float ph = phi[n];
        #pragma unroll
        for (int a = 0; a < 2; ++a) {
            const int mbase = row0 + waveM + a * 16 + g * 8;
            #pragma unroll
            for (int r = 0; r < 8; ++r) {
                // __cosf -> mul by 1/2pi + v_cos_f32 (TRANS), ~3 ops/elem
                out[(size_t)(mbase + r) * OUT_DIM + n] = __cosf(acc[a][b][r] + ph);
            }
        }
    }
}

extern "C" void kernel_launch(void* const* d_in, const int* in_sizes, int n_in,
                              void* d_out, int out_size, void* d_ws, size_t ws_size,
                              hipStream_t stream) {
    (void)in_sizes; (void)n_in; (void)out_size; (void)d_ws; (void)ws_size;
    const float* x     = (const float*)d_in[0];  // [B, IN]
    const float* theta = (const float*)d_in[1];  // [OUT, IN]
    const float* phi   = (const float*)d_in[2];  // [OUT]
    float* out = (float*)d_out;                  // [B, OUT]

    // column-blocks fastest -> the 4 blocks sharing the same x rows are
    // co-resident, so the 256MB x read is served once from HBM, rest from L2.
    dim3 grid(OUT_DIM / BN, BATCH / BM);  // (4, 1024)
    siren_ces_wmma<<<grid, 256, 0, stream>>>(x, theta, phi, out);
}
